// SchNetInteraction_76012331204796
// MI455X (gfx1250) — matrix operand order repaired and checked by reading, not compiled, not run on hardware
//
#include <hip/hip_runtime.h>
#include <hip/hip_bf16.h>

#define HID 128

typedef __attribute__((ext_vector_type(16))) __bf16 v16bf;
typedef __attribute__((ext_vector_type(8)))  float  v8f;

// ---------------------------------------------------------------------------
// WMMA fragment helpers (CDNA5 wave32 layouts, CDNA5 ISA §7.12.2)
// A (16x32 bf16): lane m = lane&15, half = lane>>4.
//   element j: vgpr v=j>>1; K = (v<4 ? 2v : 16+2(v-4)) + 8*half + (j&1)
//   -> j=0..7 : K = 8*half + j        (16 contiguous bytes -> ds_load_b128)
//   -> j=8..15: K = 16 + 8*half + (j-8)
// B (32x16 bf16): N = lane&15; element j: K = (j<8 ? 8*half + j : 16 + 8*half + (j-8)), as A
//   -> pre-swizzled in workspace so each lane's 16 elements are contiguous
// C/D (16x16 f32): element r: M = r + 8*half, N = lane&15
// ---------------------------------------------------------------------------
static __device__ __forceinline__ v16bf load_a_lds(const __bf16* __restrict__ tile,
                                                   int ldm, int m, int half, int k0) {
  v16bf a;
#pragma unroll
  for (int j = 0; j < 16; ++j) {
    int v  = j >> 1;
    int kk = ((v < 4) ? (2 * v) : (16 + 2 * (v - 4))) + half * 8 + (j & 1);
    a[j] = tile[m * ldm + k0 + kk];
  }
  return a;
}

// Swizzled weight blob: fragment (kt, nt) = 1KB: 32 lanes x 16 bf16 contiguous.
static __device__ __forceinline__ v16bf load_b_swz(const __bf16* __restrict__ Wswz,
                                                   int kt, int nt, int lane) {
  const v16bf* frag = (const v16bf*)(Wswz + (((size_t)kt * 8 + nt) << 9));
  return frag[lane];
}

#define WMMA_BF16(A, B, C) \
  __builtin_amdgcn_wmma_f32_16x16x32_bf16(false, (A), false, (B), (short)0, (C), false, false)

static __device__ __forceinline__ float silu_f32(float t) {
  // t * rcp(1+exp(-t)) : one v_exp_f32 + one v_rcp_f32 (avoid IEEE div chain)
  return t * __builtin_amdgcn_rcpf(1.0f + __expf(-t));
}

// ---------------------------------------------------------------------------
// Weight preprocessing: f32 [K,128] -> bf16 swizzled into B-fragment layout.
// dst linear index i: j = i&15, lane = (i>>4)&31, nt = (i>>9)&7, kt = i>>12
// src element: k = kt*32 + 16*(j>>3) + 8*(lane>>4) + (j&7) (zero if >= K), n = nt*16 + (lane&15)
// ---------------------------------------------------------------------------
__global__ void swizzle_w_kernel(const float* __restrict__ W, __bf16* __restrict__ out,
                                 int K, int total) {
  int i = blockIdx.x * blockDim.x + threadIdx.x;
  if (i >= total) return;
  int j    = i & 15;
  int lane = (i >> 4) & 31;
  int nt   = (i >> 9) & 7;
  int kt   = i >> 12;
  int k = kt * 32 + ((j >> 3) << 4) + ((lane >> 4) << 3) + (j & 7);
  int n = nt * 16 + (lane & 15);
  out[i] = (k < K) ? (__bf16)W[k * HID + n] : (__bf16)0.0f;
}

// x f32 -> bf16 node table (halves gather traffic; 12.8MB -> L2 resident)
__global__ void convert_x_kernel(const float* __restrict__ x, __bf16* __restrict__ xb, int n4) {
  int i = blockIdx.x * blockDim.x + threadIdx.x;
  if (i < n4) {
    const float4 v = ((const float4*)x)[i];
    __bf16* o = xb + (size_t)i * 4;
    o[0] = (__bf16)v.x; o[1] = (__bf16)v.y; o[2] = (__bf16)v.z; o[3] = (__bf16)v.w;
  }
}

__global__ void zero_f4_kernel(float4* __restrict__ p, int n4) {
  int i = blockIdx.x * blockDim.x + threadIdx.x;
  if (i < n4) p[i] = make_float4(0.f, 0.f, 0.f, 0.f);
}

// ---------------------------------------------------------------------------
// Fused edge kernel: 16 edges / block, 8 waves, each wave one 16-wide N-tile.
//   filters = silu(eattr@fW1+fb1)@fW2+fb2 ; msg = (x[src]@dW+db)*filters
//   atomicAdd into agg[dst]
// x[src] rows staged to LDS with CDNA5 async global->LDS copies.
// ---------------------------------------------------------------------------
__global__ __launch_bounds__(256)
void schnet_edge_kernel(const __bf16* __restrict__ xb,
                        const int*   __restrict__ eidx,
                        const float* __restrict__ eattr,
                        const __bf16* __restrict__ fW1s, const float* __restrict__ fb1,
                        const __bf16* __restrict__ fW2s, const float* __restrict__ fb2,
                        const __bf16* __restrict__ dWs,  const float* __restrict__ db,
                        float* __restrict__ agg, int nEdges) {
  __shared__ __align__(16) __bf16 sEat[16 * 64];
  __shared__ __align__(16) __bf16 sHid[16 * HID];
  __shared__ __align__(16) __bf16 sX[16 * HID];
  __shared__ int sSrc[16];
  __shared__ int sDst[16];

  const int tid  = threadIdx.x;
  const int lane = tid & 31;
  const int wave = tid >> 5;            // 0..7 -> N tile
  const int e0   = blockIdx.x * 16;
  const int m    = lane & 15;
  const int half = lane >> 4;
  const int colN = wave * 16 + (lane & 15);

  const int* __restrict__ src = eidx;           // edge_index row 0
  const int* __restrict__ dst = eidx + nEdges;  // edge_index row 1

  if (tid < 16) {
    int e = e0 + tid;
    if (e >= nEdges) e = nEdges - 1;
    sSrc[tid] = src[e];
    sDst[tid] = dst[e];
  }
  __syncthreads();

  // Async gather x_bf16[src] rows -> LDS. 16 rows x 256B; one 16B chunk/thread.
  {
    int r = tid >> 4;        // row 0..15
    int c = tid & 15;        // 16B chunk 0..15
    const __bf16* g = xb + (size_t)sSrc[r] * HID + c * 8;
    unsigned lds = (unsigned)(size_t)(&sX[r * HID + c * 8]);
    asm volatile("global_load_async_to_lds_b128 %0, %1, off"
                 :: "v"(lds), "v"(g) : "memory");
  }
  // edge_attr tile, K padded 50 -> 64 with zeros (f32 -> bf16 in-flight)
  for (int i = tid; i < 16 * 64; i += 256) {
    int r = i >> 6, c = i & 63;
    int e = e0 + r;
    if (e >= nEdges) e = nEdges - 1;
    sEat[i] = (c < 50) ? (__bf16)eattr[(size_t)e * 50 + c] : (__bf16)0.0f;
  }
  asm volatile("s_wait_asynccnt 0" ::: "memory");
  __syncthreads();

  // ---- filter layer 1: h = silu(eattr @ fW1 + fb1), K = 64 ----
  v8f h = {};
#pragma unroll
  for (int kt = 0; kt < 2; ++kt) {
    v16bf a = load_a_lds(sEat, 64, m, half, kt * 32);
    v16bf b = load_b_swz(fW1s, kt, wave, lane);
    h = WMMA_BF16(a, b, h);
  }
  {
    const float bias = fb1[colN];
#pragma unroll
    for (int r = 0; r < 8; ++r) {
      float t = silu_f32(h[r] + bias);
      sHid[(r + 8 * half) * HID + colN] = (__bf16)t;
    }
  }
  __syncthreads();

  // ---- filter layer 2 (h @ fW2) and message dense (x[src] @ dW), K = 128 ----
  v8f f   = {};
  v8f msg = {};
#pragma unroll
  for (int kt = 0; kt < 4; ++kt) {
    v16bf aH = load_a_lds(sHid, HID, m, half, kt * 32);
    v16bf aX = load_a_lds(sX,   HID, m, half, kt * 32);
    v16bf bF = load_b_swz(fW2s, kt, wave, lane);
    v16bf bD = load_b_swz(dWs,  kt, wave, lane);
    f   = WMMA_BF16(aH, bF, f);
    msg = WMMA_BF16(aX, bD, msg);
  }

  const float bf2 = fb2[colN];
  const float bd  = db[colN];
#pragma unroll
  for (int r = 0; r < 8; ++r) {
    int row = r + 8 * half;
    float val = (msg[r] + bd) * (f[r] + bf2);
    if (e0 + row < nEdges) {
      atomicAdd(&agg[(size_t)sDst[row] * HID + colN], val);
    }
  }
}

// ---------------------------------------------------------------------------
// Node kernel: out = x + silu(agg @ uW1 + ub1) @ uW2 + ub2
// ---------------------------------------------------------------------------
__global__ __launch_bounds__(256)
void schnet_node_kernel(const float* __restrict__ x,
                        const float* __restrict__ agg,
                        const __bf16* __restrict__ uW1s, const float* __restrict__ ub1,
                        const __bf16* __restrict__ uW2s, const float* __restrict__ ub2,
                        float* __restrict__ out, int nNodes) {
  __shared__ __align__(16) __bf16 sA[16 * HID];
  __shared__ __align__(16) __bf16 sH[16 * HID];

  const int tid  = threadIdx.x;
  const int lane = tid & 31;
  const int wave = tid >> 5;
  const int n0   = blockIdx.x * 16;
  const int m    = lane & 15;
  const int half = lane >> 4;
  const int colN = wave * 16 + (lane & 15);

  for (int i = tid; i < 16 * 32; i += 256) {
    int r  = i >> 5;
    int c4 = i & 31;
    int nr = n0 + r;
    if (nr >= nNodes) nr = nNodes - 1;
    const float4 v = ((const float4*)(agg + (size_t)nr * HID))[c4];
    __bf16* o = &sA[r * HID + c4 * 4];
    o[0] = (__bf16)v.x; o[1] = (__bf16)v.y; o[2] = (__bf16)v.z; o[3] = (__bf16)v.w;
  }
  __syncthreads();

  v8f h = {};
#pragma unroll
  for (int kt = 0; kt < 4; ++kt) {
    v16bf a = load_a_lds(sA, HID, m, half, kt * 32);
    v16bf b = load_b_swz(uW1s, kt, wave, lane);
    h = WMMA_BF16(a, b, h);
  }
  {
    const float bias = ub1[colN];
#pragma unroll
    for (int r = 0; r < 8; ++r) {
      float t = silu_f32(h[r] + bias);
      sH[(r + 8 * half) * HID + colN] = (__bf16)t;
    }
  }
  __syncthreads();

  v8f u = {};
#pragma unroll
  for (int kt = 0; kt < 4; ++kt) {
    v16bf a = load_a_lds(sH, HID, m, half, kt * 32);
    v16bf b = load_b_swz(uW2s, kt, wave, lane);
    u = WMMA_BF16(a, b, u);
  }
  const float bias2 = ub2[colN];
#pragma unroll
  for (int r = 0; r < 8; ++r) {
    int row = n0 + r + 8 * half;
    if (row < nNodes) {
      size_t off = (size_t)row * HID + colN;
      out[off] = x[off] + u[r] + bias2;
    }
  }
}

// ---------------------------------------------------------------------------
// Launch
// ---------------------------------------------------------------------------
extern "C" void kernel_launch(void* const* d_in, const int* in_sizes, int n_in,
                              void* d_out, int out_size, void* d_ws, size_t ws_size,
                              hipStream_t stream) {
  const float* x     = (const float*)d_in[0];
  const int*   eidx  = (const int*)  d_in[1];
  const float* eattr = (const float*)d_in[2];
  const float* fW1   = (const float*)d_in[3];
  const float* fb1   = (const float*)d_in[4];
  const float* fW2   = (const float*)d_in[5];
  const float* fb2   = (const float*)d_in[6];
  const float* dW    = (const float*)d_in[7];
  const float* db    = (const float*)d_in[8];
  const float* uW1   = (const float*)d_in[9];
  const float* ub1   = (const float*)d_in[10];
  const float* uW2   = (const float*)d_in[11];
  const float* ub2   = (const float*)d_in[12];
  float* out = (float*)d_out;

  const int nNodes = in_sizes[0] / HID;
  const int nEdges = in_sizes[1] / 2;

  // ws: [agg f32 N*H][xb bf16 N*H][fW1s 2kt][fW2s 4kt][dWs 4kt][uW1s 4kt][uW2s 4kt]
  const int SW1 = 2 * 8 * 512;   // swizzled elements, Kpad=64
  const int SW  = 4 * 8 * 512;   // swizzled elements, K=128
  float* agg = (float*)d_ws;
  char* p = (char*)d_ws + (size_t)nNodes * HID * sizeof(float);
  __bf16* xb   = (__bf16*)p; p += (size_t)nNodes * HID * sizeof(__bf16);
  __bf16* fW1s = (__bf16*)p; p += (size_t)SW1 * sizeof(__bf16);
  __bf16* fW2s = (__bf16*)p; p += (size_t)SW * sizeof(__bf16);
  __bf16* dWs  = (__bf16*)p; p += (size_t)SW * sizeof(__bf16);
  __bf16* uW1s = (__bf16*)p; p += (size_t)SW * sizeof(__bf16);
  __bf16* uW2s = (__bf16*)p; p += (size_t)SW * sizeof(__bf16);

  swizzle_w_kernel<<<(SW1 + 255) / 256, 256, 0, stream>>>(fW1, fW1s, 50, SW1);
  swizzle_w_kernel<<<(SW + 255) / 256, 256, 0, stream>>>(fW2, fW2s, HID, SW);
  swizzle_w_kernel<<<(SW + 255) / 256, 256, 0, stream>>>(dW,  dWs,  HID, SW);
  swizzle_w_kernel<<<(SW + 255) / 256, 256, 0, stream>>>(uW1, uW1s, HID, SW);
  swizzle_w_kernel<<<(SW + 255) / 256, 256, 0, stream>>>(uW2, uW2s, HID, SW);

  const int n4 = nNodes * HID / 4;
  convert_x_kernel<<<(n4 + 255) / 256, 256, 0, stream>>>(x, xb, n4);
  zero_f4_kernel<<<(n4 + 255) / 256, 256, 0, stream>>>((float4*)agg, n4);

  schnet_edge_kernel<<<(nEdges + 15) / 16, 256, 0, stream>>>(
      xb, eidx, eattr, fW1s, fb1, fW2s, fb2, dWs, db, agg, nEdges);

  schnet_node_kernel<<<(nNodes + 15) / 16, 256, 0, stream>>>(
      x, agg, uW1s, ub1, uW2s, ub2, out, nNodes);
}
